// GNN_29832842838748
// MI455X (gfx1250) — matrix operand order repaired and checked by reading, not compiled, not run on hardware
//
#include <hip/hip_runtime.h>
#include <math.h>

typedef _Float16 v16h __attribute__((ext_vector_type(16)));
typedef _Float16 v8h  __attribute__((ext_vector_type(8)));
typedef float    v8f  __attribute__((ext_vector_type(8)));
typedef float    v2f  __attribute__((ext_vector_type(2)));

#define KNN_K 8
#define HAVE_WMMA_X4 __has_builtin(__builtin_amdgcn_wmma_f32_16x16x4_f32)

// ---------------------------------------------------------------- utilities

__device__ __forceinline__ void atomic_max_float(float* addr, float val) {
  // int-max for non-negative patterns, uint-min for negative patterns.
  // Converges to true float max from a -inf (0xFF800000) initialization.
  if (val >= 0.0f) atomicMax((int*)addr, __float_as_int(val));
  else             atomicMin((unsigned int*)addr, __float_as_uint(val));
}

__device__ __forceinline__ void topk_insert(float (&bd)[KNN_K], int (&bi)[KNN_K],
                                            float& worst, float s, int cid) {
  if (s < worst) {
    int ws = 0; float w = bd[0];
#pragma unroll
    for (int i = 1; i < KNN_K; ++i) { if (bd[i] > w) { w = bd[i]; ws = i; } }
#pragma unroll
    for (int i = 0; i < KNN_K; ++i) { if (i == ws) { bd[i] = s; bi[i] = cid; } }
    w = bd[0];
#pragma unroll
    for (int i = 1; i < KNN_K; ++i) w = fmaxf(w, bd[i]);
    worst = w;
  }
}

__global__ void fill_kernel(float* __restrict__ p, float v, int n) {
  int i = blockIdx.x * blockDim.x + threadIdx.x;
  if (i < n) p[i] = v;
}

// ---------------------------------------------------------------- kNN
// One wave = 16 queries (query n = lane&15, duplicated in both lane halves).
// Candidates staged in LDS as (-2x,-2y,-2z,|p|^2). Score tile via f32 WMMA:
//   A(16 cand x 4) = (-2p, |p|^2),  B(4 x 16 queries) = (q, 1)
//   C[m][n] = -2 p.q + |p|^2 = d^2 - |q|^2   (ranking-equivalent to d^2)
// Lane half 0 sees candidate rows m=0..7 of each tile, half 1 sees m=8..15;
// per-lane top-8 halves are merged through LDS at the end.
// Tiles are processed in batches of 4 (loads+WMMAs issued together), and a
// per-tile fmin-tree prefilter keeps the insert machinery off the hot path.

__global__ void __launch_bounds__(256)
knn_kernel(const float* __restrict__ coord, int* __restrict__ nbr, int N) {
  const int tid  = threadIdx.x;
  const int wave = tid >> 5;
  const int lane = tid & 31;
  const int half = lane >> 4;
  const int l16  = lane & 15;

  const int q_log = blockIdx.x * 128 + wave * 16 + l16;
  const int q = (q_log < N) ? q_log : (N - 1);
  const float qx = coord[3*q+0], qy = coord[3*q+1], qz = coord[3*q+2];

#if HAVE_WMMA_X4
  // B layout (4x16 f32): lanes 0-15 hold K0,K1 (vgpr0,1); lanes 16-31 K2,K3.
  v2f bq;
  bq.x = half ? qz   : qx;
  bq.y = half ? 1.0f : qy;
#endif

  __shared__ __align__(16) float cand[256][4];
  __shared__ float mD[8][16][16];
  __shared__ int   mI[8][16][16];

  float bd[KNN_K]; int bi[KNN_K];
#pragma unroll
  for (int i = 0; i < KNN_K; ++i) { bd[i] = 3.0e38f; bi[i] = q; }
  float worst = 3.0e38f;

  for (int cbase = 0; cbase < N; cbase += 256) {
    __syncthreads();
    {
      const int ci = cbase + tid;
      float px = 0.f, py = 0.f, pz = 0.f, pw = 3.0e38f;  // pad -> huge score
      if (ci < N) {
        px = coord[3*ci+0]; py = coord[3*ci+1]; pz = coord[3*ci+2];
        pw = px*px + py*py + pz*pz;
      }
      cand[tid][0] = -2.0f*px; cand[tid][1] = -2.0f*py;
      cand[tid][2] = -2.0f*pz; cand[tid][3] = pw;
    }
    __syncthreads();

#if HAVE_WMMA_X4
    for (int tb = 0; tb < 16; tb += 4) {
      // Batch: 4 A-fragment LDS loads, then 4 WMMAs (single dscnt wait,
      // hazard gaps filled by the independent issues).
      v2f a[4];
#pragma unroll
      for (int u = 0; u < 4; ++u)
        a[u] = *(const v2f*)&cand[(tb + u)*16 + l16][half*2];
      v8f c[4];
#pragma unroll
      for (int u = 0; u < 4; ++u) {
        v8f z = {};
        c[u] = __builtin_amdgcn_wmma_f32_16x16x4_f32(false, a[u], false, bq,
                                                     (short)0, z, false, false);
      }
#pragma unroll
      for (int u = 0; u < 4; ++u) {
        const int mbase = cbase + (tb + u)*16 + half*8;
        // v_min3-friendly reduction: one branch per 8 scores.
        const float mn =
            fminf(fminf(fminf(c[u][0], c[u][1]), fminf(c[u][2], c[u][3])),
                  fminf(fminf(c[u][4], c[u][5]), fminf(c[u][6], c[u][7])));
        if (mn < worst) {
#pragma unroll
          for (int r = 0; r < 8; ++r)
            topk_insert(bd, bi, worst, c[u][r], mbase + r);
        }
      }
    }
#else
    for (int t = 0; t < 16; ++t) {
      float s8[8];
#pragma unroll
      for (int r = 0; r < 8; ++r) {
        const int mm = t*16 + half*8 + r;
        s8[r] = fmaf(qx, cand[mm][0],
                fmaf(qy, cand[mm][1],
                fmaf(qz, cand[mm][2], cand[mm][3])));
      }
      const float mn = fminf(fminf(fminf(s8[0], s8[1]), fminf(s8[2], s8[3])),
                             fminf(fminf(s8[4], s8[5]), fminf(s8[6], s8[7])));
      if (mn < worst) {
#pragma unroll
        for (int r = 0; r < 8; ++r)
          topk_insert(bd, bi, worst, s8[r], cbase + t*16 + half*8 + r);
      }
    }
#endif
  }

  __syncthreads();
#pragma unroll
  for (int i = 0; i < KNN_K; ++i) {
    mD[wave][l16][half*8 + i] = bd[i];
    mI[wave][l16][half*8 + i] = bi[i];
  }
  __syncthreads();
  if (half == 0 && q_log < N) {
#pragma unroll
    for (int i = 0; i < KNN_K; ++i)
      topk_insert(bd, bi, worst, mD[wave][l16][8 + i], mI[wave][l16][8 + i]);
#pragma unroll
    for (int i = 0; i < KNN_K; ++i) nbr[q_log*KNN_K + i] = bi[i];
  }
}

// ---------------------------------------------------------------- EdgeConv
// One wave = one 16-edge tile.  e = tile*16 + m; src = e/8; dst = nbr[e].
// Edge features [x_dst, x_src - x_dst] (K1 = pad32(2*CI)) staged as f16 in
// LDS; GEMM1 (K1 -> CO) and GEMM2 (CO -> CO) via v_wmma_f32_16x16x32_f16,
// bias+ReLU between; results scatter-max'ed into xout[dst][*].

template <int CI, int CO, bool RELU_IN>
__global__ void __launch_bounds__(128)
edgeconv_kernel(const float* __restrict__ xin, const int* __restrict__ nbr,
                const float* __restrict__ W1, const float* __restrict__ B1,
                const float* __restrict__ W2, const float* __restrict__ B2,
                float* __restrict__ xout, int N) {
  constexpr int K1  = (2*CI + 31) & ~31;   // GEMM1 K, padded to 32
  constexpr int COP = (CO + 15) & ~15;     // padded output channels
  constexpr int K2  = (COP + 31) & ~31;    // GEMM2 K, padded to 32
  constexpr int NT  = COP / 16;
  constexpr int KC1 = K1 / 32;
  constexpr int KC2 = K2 / 32;
  constexpr int WAVES = 4;

  __shared__ __align__(16) _Float16 W1T[COP][K1];   // W1 transposed: [n][k]
  __shared__ __align__(16) _Float16 W2T[COP][K2];
  __shared__ float b1s[COP], b2s[COP];
  __shared__ __align__(16) _Float16 EF[WAVES][16][K1];  // edge features
  __shared__ __align__(16) _Float16 Hs[WAVES][16][K2];  // hidden activations
  __shared__ int dsts[WAVES][16];

  const int tid  = threadIdx.x;
  const int wave = tid >> 5;
  const int lane = tid & 31;
  const int half = lane >> 4;
  const int l16  = lane & 15;
  const int E    = N * KNN_K;

  // --- stage (transposed, f16, zero-padded) weights + biases in LDS
  for (int i = tid; i < COP*K1; i += 128) {
    const int n = i / K1, k = i % K1;
    W1T[n][k] = (_Float16)((n < CO && k < 2*CI) ? W1[k*CO + n] : 0.0f);
  }
  for (int i = tid; i < COP*K2; i += 128) {
    const int n = i / K2, k = i % K2;
    W2T[n][k] = (_Float16)((n < CO && k < CO) ? W2[k*CO + n] : 0.0f);
  }
  for (int i = tid; i < COP; i += 128) {
    b1s[i] = (i < CO) ? B1[i] : 0.0f;
    b2s[i] = (i < CO) ? B2[i] : 0.0f;
  }
  __syncthreads();

  // --- gather edge-feature tile
  const int e0 = (blockIdx.x * WAVES + wave) * 16;
  const int e  = (e0 + l16 < E) ? (e0 + l16) : (E - 1);
  const int sn = e >> 3;          // src = e / k
  const int dn = nbr[e];          // dst
  if (half == 0) dsts[wave][l16] = dn;

  constexpr int CH = CI / 2;      // channels handled per lane-half
  const int c0 = half * CH;
#pragma unroll
  for (int c = 0; c < CH; ++c) {
    float xd = xin[dn*CI + c0 + c];
    float xs = xin[sn*CI + c0 + c];
    if (RELU_IN) { xd = fmaxf(xd, 0.0f); xs = fmaxf(xs, 0.0f); }
    EF[wave][l16][c0 + c]      = (_Float16)xd;
    EF[wave][l16][CI + c0 + c] = (_Float16)(xs - xd);
  }
  if (K1 > 2*CI) {
    for (int j = lane; j < 16*(K1 - 2*CI); j += 32) {
      const int m = j / (K1 - 2*CI), k = 2*CI + j % (K1 - 2*CI);
      EF[wave][m][k] = (_Float16)0.0f;
    }
  }
  __syncthreads();

  // --- GEMM1: [16 x K1] x [K1 x COP]
  v8f acc1[NT] = {};
#pragma unroll
  for (int kc = 0; kc < KC1; ++kc) {
    // A (16x32 f16): lanes 0-15: K=kc*32+{0..7, 16..23}; lanes 16-31: +8.
    const _Float16* ap = &EF[wave][l16][kc*32 + half*8];
    v16h a;
    *((v8h*)&a + 0) = *(const v8h*)(ap);
    *((v8h*)&a + 1) = *(const v8h*)(ap + 16);
#pragma unroll
    for (int nt = 0; nt < NT; ++nt) {
      // B (32x16 f16): lane n=lane&15; lanes 0-15: K=kc*32+{0..7, 16..23}; lanes 16-31: +8.
      const _Float16* bp = &W1T[nt*16 + l16][kc*32 + half*8];
      v16h b;
      *((v8h*)&b + 0) = *(const v8h*)(bp);
      *((v8h*)&b + 1) = *(const v8h*)(bp + 16);
      acc1[nt] = __builtin_amdgcn_wmma_f32_16x16x32_f16(
          false, a, false, b, (short)0, acc1[nt], false, false);
    }
  }

  // --- bias + ReLU -> Hs (f16).  C layout: lane holds col n=lane&15,
  // rows m = half*8 + r.
#pragma unroll
  for (int nt = 0; nt < NT; ++nt) {
    const int n = nt*16 + l16;
    const float bb = b1s[n];
#pragma unroll
    for (int r = 0; r < 8; ++r) {
      const int m = half*8 + r;
      Hs[wave][m][n] = (_Float16)fmaxf(acc1[nt][r] + bb, 0.0f);
    }
  }
  if (K2 > COP) {
    for (int j = lane; j < 16*(K2 - COP); j += 32) {
      const int m = j / (K2 - COP), k = COP + j % (K2 - COP);
      Hs[wave][m][k] = (_Float16)0.0f;
    }
  }
  __syncthreads();

  // --- GEMM2: [16 x K2] x [K2 x COP]
  v8f acc2[NT] = {};
#pragma unroll
  for (int kc = 0; kc < KC2; ++kc) {
    const _Float16* ap = &Hs[wave][l16][kc*32 + half*8];
    v16h a;
    *((v8h*)&a + 0) = *(const v8h*)(ap);
    *((v8h*)&a + 1) = *(const v8h*)(ap + 16);
#pragma unroll
    for (int nt = 0; nt < NT; ++nt) {
      const _Float16* bp = &W2T[nt*16 + l16][kc*32 + half*8];
      v16h b;
      *((v8h*)&b + 0) = *(const v8h*)(bp);
      *((v8h*)&b + 1) = *(const v8h*)(bp + 16);
      acc2[nt] = __builtin_amdgcn_wmma_f32_16x16x32_f16(
          false, a, false, b, (short)0, acc2[nt], false, false);
    }
  }

  // --- bias + scatter segment-max over dst
#pragma unroll
  for (int nt = 0; nt < NT; ++nt) {
    const int n = nt*16 + l16;
    const float bb = b2s[n];
#pragma unroll
    for (int r = 0; r < 8; ++r) {
      const int m = half*8 + r;
      if (n < CO && (e0 + m) < E)
        atomic_max_float(&xout[(size_t)dsts[wave][m]*CO + n], acc2[nt][r] + bb);
    }
  }
}

// ---------------------------------------------------------------- launch

extern "C" void kernel_launch(void* const* d_in, const int* in_sizes, int n_in,
                              void* d_out, int out_size, void* d_ws, size_t ws_size,
                              hipStream_t stream) {
  (void)out_size; (void)ws_size;
  const int N = in_sizes[0] / 3;   // coord1 is [N,3]
  const int E = N * KNN_K;

  const float* coord[2] = { (const float*)d_in[0], (const float*)d_in[3] };
  const float* feat[2]  = { (const float*)d_in[1], (const float*)d_in[4] };

  // params_x / params_y: 12 leaves each (W1,b1,W2,b2 per layer) if the
  // harness flattens pytrees; otherwise a packed fp32 blob per branch.
  const float* P[2][12];
  if (n_in >= 30) {
    for (int b = 0; b < 2; ++b)
      for (int i = 0; i < 12; ++i)
        P[b][i] = (const float*)d_in[6 + b*12 + i];
  } else {
    static const size_t off[12] = {0,512,576,4672, 4736,12928,12992,17088,
                                   17152,17536,17539,17548};
    for (int b = 0; b < 2; ++b) {
      const float* base = (const float*)d_in[6 + b];
      for (int i = 0; i < 12; ++i) P[b][i] = base + off[i];
    }
  }

  char* ws = (char*)d_ws;
  int*   nbr = (int*)ws;        ws += (size_t)E * sizeof(int);
  float* xa  = (float*)ws;      ws += (size_t)N * 64 * sizeof(float);
  float* xb  = (float*)ws;      ws += (size_t)N * 64 * sizeof(float);

  const float NEG_INF = -__builtin_huge_valf();
  float* out = (float*)d_out;
  const int convWG = (E + 63) / 64;   // 4 waves x 16 edges per WG

  for (int b = 0; b < 2; ++b) {
    knn_kernel<<<(N + 127)/128, 256, 0, stream>>>(coord[b], nbr, N);

    fill_kernel<<<(N*64 + 255)/256, 256, 0, stream>>>(xa, NEG_INF, N*64);
    edgeconv_kernel<4, 64, false><<<convWG, 128, 0, stream>>>(
        feat[b], nbr, P[b][0], P[b][1], P[b][2], P[b][3], xa, N);

    fill_kernel<<<(N*64 + 255)/256, 256, 0, stream>>>(xb, NEG_INF, N*64);
    edgeconv_kernel<64, 64, true><<<convWG, 128, 0, stream>>>(
        xa, nbr, P[b][4], P[b][5], P[b][6], P[b][7], xb, N);

    float* outb = out + (size_t)b * N * 3;
    fill_kernel<<<(N*3 + 255)/256, 256, 0, stream>>>(outb, NEG_INF, N*3);
    edgeconv_kernel<64, 3, true><<<convWG, 128, 0, stream>>>(
        xb, nbr, P[b][8], P[b][9], P[b][10], P[b][11], outb, N);
  }
}
